// BidirectionalAttentionV2_61933428408466
// MI455X (gfx1250) — matrix operand order repaired and checked by reading, not compiled, not run on hardware
//
#include <hip/hip_runtime.h>
#include <hip/hip_bf16.h>
#include <cstdint>

// ---------- problem constants ----------
#define BB 8
#define HH 12
#define TT 1024
#define NN 256          // head dim
#define BH (BB*HH)      // 96
#define TM 128          // query rows per block (8 waves x 16)
#define TS 64           // key rows per tile
#define KP 264          // K tile pitch (halfs): 256 + 8 pad  -> 528B rows, 16B aligned
#define VP 72           // V tile pitch (halfs): 64 + 8 pad   -> 144B rows, 16B aligned
#define PP 72           // P scratch pitch (halfs)
#define SCALE 0.0625f   // 1/sqrt(256)
#define TWO_PI 6.28318530717958647692f

typedef __attribute__((ext_vector_type(16))) _Float16 v16h;
typedef __attribute__((ext_vector_type(8)))  _Float16 v8h;
typedef __attribute__((ext_vector_type(4)))  _Float16 v4h;
typedef __attribute__((ext_vector_type(8)))  float    v8f;

// Load a 16-f16 WMMA fragment as two contiguous 16B chunks (LDS -> ds_load_b128 x2)
__device__ __forceinline__ v16h ldfrag(const _Float16* p0, const _Float16* p1) {
    v8h a = *(const v8h*)p0;
    v8h b = *(const v8h*)p1;
    return __builtin_shufflevector(a, b, 0,1,2,3,4,5,6,7,8,9,10,11,12,13,14,15);
}

// Async 16B global->LDS copy (ASYNCcnt-tracked, gfx1250)
__device__ __forceinline__ void async_b128(uint32_t lds_addr, const void* gptr) {
    asm volatile("global_load_async_to_lds_b128 %0, %1, off"
                 :: "v"(lds_addr), "v"(gptr) : "memory");
}
__device__ __forceinline__ void wait_async0() {
    asm volatile("s_wait_asynccnt 0" ::: "memory");
}

// RoPE for an (even,odd) pair; ph_j = fract(t * freqs[j%128]) * 2pi (matches reference)
__device__ __forceinline__ void rope2(float t, int k_even, float ve, float vo,
                                      const float* __restrict__ fr,
                                      float& oe, float& oo) {
    const int je = k_even & 127;
    const int jo = (k_even + 1) & 127;
    float phe = t * fr[je]; phe -= floorf(phe);
    float pho = t * fr[jo]; pho -= floorf(pho);
    float se, ce, so, co;
    __sincosf(phe * TWO_PI, &se, &ce);
    __sincosf(pho * TWO_PI, &so, &co);
    oe = ve * ce - vo * se;
    oo = vo * co + ve * so;
}

// =====================================================================
// Prep kernel A: QR = f16(RoPE(Q))  laid out [bh][t][k] row-major
// =====================================================================
__global__ __launch_bounds__(256, 1)
void rope_prep_kernel(const float* __restrict__ Qg, const float* __restrict__ Fg,
                      _Float16* __restrict__ QR) {
    __shared__ float freq_s[128];
    const int tid = threadIdx.x;
    if (tid < 128) freq_s[tid] = Fg[tid];
    __syncthreads();

    const int bh = blockIdx.x >> 4;
    const int s0 = (blockIdx.x & 15) * TS;
    const int r  = tid >> 2;
    const int kb = (tid & 3) * 64;
    const float t = (float)(s0 + r);
    const float* qrow = Qg + ((size_t)(bh * TT + s0 + r)) * NN;
    _Float16*    orow = QR + ((size_t)(bh * TT + s0 + r)) * NN;
    #pragma unroll
    for (int i = 0; i < 16; ++i) {
        const int k = kb + 4 * i;
        float4 q = *(const float4*)(qrow + k);
        float o0, o1, o2, o3;
        rope2(t, k,     q.x, q.y, freq_s, o0, o1);
        rope2(t, k + 2, q.z, q.w, freq_s, o2, o3);
        v4h hv = { (_Float16)o0, (_Float16)o1, (_Float16)o2, (_Float16)o3 };
        *(v4h*)(orow + k) = hv;
    }
}

// =====================================================================
// Prep kernel B: VT = f16(V)^T  laid out [bh][n][t]  (coalesced via LDS tile)
// =====================================================================
__global__ __launch_bounds__(256, 1)
void vtrans_prep_kernel(const float* __restrict__ Vg, _Float16* __restrict__ VT) {
    __shared__ _Float16 tile[NN * VP];
    const int tid  = threadIdx.x;
    const int lane = tid & 31;
    const int wv   = tid >> 5;
    const int bh = blockIdx.x >> 4;
    const int s0 = (blockIdx.x & 15) * TS;

    {
        const int r  = tid >> 2;
        const int nb = (tid & 3) * 64;
        const float* vrow = Vg + ((size_t)(bh * TT + s0 + r)) * NN;
        #pragma unroll
        for (int i = 0; i < 16; ++i) {
            const int n = nb + 4 * i;
            float4 v = *(const float4*)(vrow + n);
            tile[(n + 0) * VP + r] = (_Float16)v.x;
            tile[(n + 1) * VP + r] = (_Float16)v.y;
            tile[(n + 2) * VP + r] = (_Float16)v.z;
            tile[(n + 3) * VP + r] = (_Float16)v.w;
        }
    }
    __syncthreads();

    #pragma unroll
    for (int p = 0; p < 8; ++p) {
        const int n    = p * 32 + wv * 4 + (lane >> 3);
        const int soff = (lane & 7) * 8;
        v8h d = *(const v8h*)&tile[n * VP + soff];
        *(v8h*)(VT + ((size_t)(bh * NN + n)) * TT + s0 + soff) = d;
    }
}

// =====================================================================
// Main FA kernel (split path): f16 QR / VT inputs, async double-buffered tiles
// =====================================================================
__device__ __forceinline__ void issue_tile_async(const _Float16* __restrict__ QR,
                                                 const _Float16* __restrict__ VT,
                                                 int bh, int s0,
                                                 _Float16* kbuf, _Float16* vbuf, int tid) {
    {
        const int r  = tid >> 2;
        const int cb = (tid & 3) * 8;
        const _Float16* g = QR + ((size_t)(bh * TT + s0 + r)) * NN + cb * 8;
        const uint32_t  l = (uint32_t)(uintptr_t)(kbuf + r * KP + cb * 8);
        #pragma unroll
        for (int i = 0; i < 8; ++i) async_b128(l + i * 16, g + i * 8);
    }
    {
        const _Float16* g = VT + ((size_t)(bh * NN + tid)) * TT + s0;
        const uint32_t  l = (uint32_t)(uintptr_t)(vbuf + tid * VP);
        #pragma unroll
        for (int i = 0; i < 8; ++i) async_b128(l + i * 16, g + i * 8);
    }
}

__global__ __launch_bounds__(256, 1)
void fa2_kernel(const _Float16* __restrict__ QR, const _Float16* __restrict__ VT,
                float* __restrict__ Og) {
    __shared__ _Float16 Kbuf[2][TS * KP];
    __shared__ _Float16 Vbuf[2][NN * VP];
    __shared__ _Float16 Plds[8 * 16 * PP];

    const int tid  = threadIdx.x;
    const int lane = tid & 31;
    const int wv   = tid >> 5;
    const int h    = lane >> 4;
    const int nl   = lane & 15;
    const int bh   = blockIdx.x >> 3;
    const int rb   = blockIdx.x & 7;

    const int rowG = rb * TM + wv * 16 + nl;
    const _Float16* qrow = QR + ((size_t)(bh * TT + rowG)) * NN;
    v16h aq[8];
    #pragma unroll
    for (int f = 0; f < 8; ++f) {
        const _Float16* p = qrow + 32 * f + 8 * h;
        aq[f] = ldfrag(p, p + 16);
    }

    float m_run[8], l_run[8];
    #pragma unroll
    for (int r = 0; r < 8; ++r) { m_run[r] = -3.0e38f; l_run[r] = 0.0f; }
    v8f o[16];
    #pragma unroll
    for (int nt = 0; nt < 16; ++nt) { v8f z = {}; o[nt] = z; }

    _Float16* Pw = Plds + wv * 16 * PP;

    issue_tile_async(QR, VT, bh, 0, Kbuf[0], Vbuf[0], tid);

    for (int st = 0; st < TT / TS; ++st) {
        const int cur = st & 1;
        wait_async0();
        __syncthreads();
        if (st < TT / TS - 1)
            issue_tile_async(QR, VT, bh, (st + 1) * TS, Kbuf[cur ^ 1], Vbuf[cur ^ 1], tid);

        const _Float16* K0 = Kbuf[cur];
        const _Float16* V0 = Vbuf[cur];

        // ---- S strip: per s-tile, batch-load all 8 B frags, then 8 WMMAs ----
        v8f c[4];
        #pragma unroll
        for (int jt = 0; jt < 4; ++jt) {
            const _Float16* krow = &K0[(jt * 16 + nl) * KP + 8 * h];
            v16h bk[8];
            #pragma unroll
            for (int f = 0; f < 8; ++f)
                bk[f] = ldfrag(krow + 32 * f, krow + 32 * f + 16);
            v8f cj = {};
            #pragma unroll
            for (int f = 0; f < 8; ++f)
                cj = __builtin_amdgcn_wmma_f32_16x16x32_f16(
                        false, aq[f], false, bk[f], (short)0, cj, false, false);
            c[jt] = cj;
        }

        // ---- online softmax ----
        float fscale[8], mnew[8];
        #pragma unroll
        for (int jt = 0; jt < 4; ++jt)
            #pragma unroll
            for (int r = 0; r < 8; ++r) c[jt][r] *= SCALE;

        #pragma unroll
        for (int r = 0; r < 8; ++r) {
            float v = fmaxf(fmaxf(c[0][r], c[1][r]), fmaxf(c[2][r], c[3][r]));
            v = fmaxf(v, __shfl_xor(v, 1, 32));
            v = fmaxf(v, __shfl_xor(v, 2, 32));
            v = fmaxf(v, __shfl_xor(v, 4, 32));
            v = fmaxf(v, __shfl_xor(v, 8, 32));
            mnew[r]   = fmaxf(m_run[r], v);
            fscale[r] = __expf(m_run[r] - mnew[r]);
            m_run[r]  = mnew[r];
        }

        #pragma unroll
        for (int r = 0; r < 8; ++r) {
            float ps = 0.0f;
            #pragma unroll
            for (int jt = 0; jt < 4; ++jt) {
                float p = __expf(c[jt][r] - mnew[r]);
                ps += p;
                Pw[(h * 8 + r) * PP + jt * 16 + nl] = (_Float16)p;
            }
            ps += __shfl_xor(ps, 1, 32);
            ps += __shfl_xor(ps, 2, 32);
            ps += __shfl_xor(ps, 4, 32);
            ps += __shfl_xor(ps, 8, 32);
            l_run[r] = l_run[r] * fscale[r] + ps;
        }

        #pragma unroll
        for (int nt = 0; nt < 16; ++nt)
            #pragma unroll
            for (int r = 0; r < 8; ++r) o[nt][r] *= fscale[r];

        // ---- O += P x V: groups of 4 n-tiles, batch 8 frags then 8 WMMAs ----
        const _Float16* pr = Pw + nl * PP;
        v16h pa0 = ldfrag(pr + 8 * h,      pr + 16 + 8 * h);
        v16h pa1 = ldfrag(pr + 32 + 8 * h, pr + 48 + 8 * h);
        #pragma unroll
        for (int g = 0; g < 4; ++g) {
            v16h bv[8];
            #pragma unroll
            for (int i = 0; i < 4; ++i) {
                const _Float16* vb = &V0[((g * 4 + i) * 16 + nl) * VP];
                bv[2 * i]     = ldfrag(vb + 8 * h,      vb + 16 + 8 * h);
                bv[2 * i + 1] = ldfrag(vb + 32 + 8 * h, vb + 48 + 8 * h);
            }
            #pragma unroll
            for (int i = 0; i < 4; ++i) {
                const int nt = g * 4 + i;
                o[nt] = __builtin_amdgcn_wmma_f32_16x16x32_f16(
                            false, pa0, false, bv[2 * i], (short)0, o[nt], false, false);
                o[nt] = __builtin_amdgcn_wmma_f32_16x16x32_f16(
                            false, pa1, false, bv[2 * i + 1], (short)0, o[nt], false, false);
            }
        }
    }

    #pragma unroll
    for (int r = 0; r < 8; ++r) l_run[r] = 1.0f / l_run[r];
    #pragma unroll
    for (int nt = 0; nt < 16; ++nt) {
        #pragma unroll
        for (int r = 0; r < 8; ++r) {
            const int row = rb * TM + wv * 16 + h * 8 + r;
            Og[((size_t)(bh * TT + row)) * NN + nt * 16 + nl] = o[nt][r] * l_run[r];
        }
    }
}

// =====================================================================
// Fallback: fused single kernel (used only if ws too small)
// =====================================================================
__global__ __launch_bounds__(256, 1)
void fa_fused_kernel(const float* __restrict__ Qg, const float* __restrict__ Vg,
                     const float* __restrict__ Fg, float* __restrict__ Og) {
    __shared__ float    freq_s[128];
    __shared__ _Float16 Klds[TS * KP];
    __shared__ _Float16 Vlds[NN * VP];
    __shared__ _Float16 Plds[8 * 16 * PP];

    const int tid  = threadIdx.x;
    const int lane = tid & 31;
    const int wv   = tid >> 5;
    const int h    = lane >> 4;
    const int nl   = lane & 15;
    const int bh   = blockIdx.x >> 3;
    const int rb   = blockIdx.x & 7;

    if (tid < 128) freq_s[tid] = Fg[tid];
    __syncthreads();

    const int rowG = rb * TM + wv * 16 + nl;
    const float tq = (float)rowG;
    const float* qrow = Qg + ((size_t)(bh * TT + rowG)) * NN;

    v16h aq[8];
    #pragma unroll
    for (int f = 0; f < 8; ++f) {
        const int k1 = 32 * f + 8 * h;
        const int k2 = k1 + 16;
        float4 a0 = *(const float4*)(qrow + k1);
        float4 a1 = *(const float4*)(qrow + k1 + 4);
        float4 b0 = *(const float4*)(qrow + k2);
        float4 b1 = *(const float4*)(qrow + k2 + 4);
        float e[16];
        rope2(tq, k1,     a0.x, a0.y, freq_s, e[0],  e[1]);
        rope2(tq, k1 + 2, a0.z, a0.w, freq_s, e[2],  e[3]);
        rope2(tq, k1 + 4, a1.x, a1.y, freq_s, e[4],  e[5]);
        rope2(tq, k1 + 6, a1.z, a1.w, freq_s, e[6],  e[7]);
        rope2(tq, k2,     b0.x, b0.y, freq_s, e[8],  e[9]);
        rope2(tq, k2 + 2, b0.z, b0.w, freq_s, e[10], e[11]);
        rope2(tq, k2 + 4, b1.x, b1.y, freq_s, e[12], e[13]);
        rope2(tq, k2 + 6, b1.z, b1.w, freq_s, e[14], e[15]);
        v16h v;
        #pragma unroll
        for (int i = 0; i < 16; ++i) v[i] = (_Float16)e[i];
        aq[f] = v;
    }

    float m_run[8], l_run[8];
    #pragma unroll
    for (int r = 0; r < 8; ++r) { m_run[r] = -3.0e38f; l_run[r] = 0.0f; }
    v8f o[16];
    #pragma unroll
    for (int nt = 0; nt < 16; ++nt) { v8f z = {}; o[nt] = z; }

    _Float16* Pw = Plds + wv * 16 * PP;

    for (int st = 0; st < TT / TS; ++st) {
        const int s0 = st * TS;
        __syncthreads();
        {
            const int r  = tid >> 2;
            const int kb = (tid & 3) * 64;
            const float tk = (float)(s0 + r);
            const float* krow = Qg + ((size_t)(bh * TT + s0 + r)) * NN;
            #pragma unroll
            for (int i = 0; i < 16; ++i) {
                const int k = kb + 4 * i;
                float4 q = *(const float4*)(krow + k);
                float o0, o1, o2, o3;
                rope2(tk, k,     q.x, q.y, freq_s, o0, o1);
                rope2(tk, k + 2, q.z, q.w, freq_s, o2, o3);
                v4h hv = { (_Float16)o0, (_Float16)o1, (_Float16)o2, (_Float16)o3 };
                *(v4h*)&Klds[r * KP + k] = hv;
            }
        }
        {
            const int r  = tid >> 2;
            const int nb = (tid & 3) * 64;
            const float* vrow = Vg + ((size_t)(bh * TT + s0 + r)) * NN;
            #pragma unroll
            for (int i = 0; i < 16; ++i) {
                const int n = nb + 4 * i;
                float4 v = *(const float4*)(vrow + n);
                Vlds[(n + 0) * VP + r] = (_Float16)v.x;
                Vlds[(n + 1) * VP + r] = (_Float16)v.y;
                Vlds[(n + 2) * VP + r] = (_Float16)v.z;
                Vlds[(n + 3) * VP + r] = (_Float16)v.w;
            }
        }
        __syncthreads();

        v8f c[4];
        #pragma unroll
        for (int jt = 0; jt < 4; ++jt) {
            const _Float16* krow = &Klds[(jt * 16 + nl) * KP + 8 * h];
            v16h bk[8];
            #pragma unroll
            for (int f = 0; f < 8; ++f)
                bk[f] = ldfrag(krow + 32 * f, krow + 32 * f + 16);
            v8f cj = {};
            #pragma unroll
            for (int f = 0; f < 8; ++f)
                cj = __builtin_amdgcn_wmma_f32_16x16x32_f16(
                        false, aq[f], false, bk[f], (short)0, cj, false, false);
            c[jt] = cj;
        }

        float fscale[8], mnew[8];
        #pragma unroll
        for (int jt = 0; jt < 4; ++jt)
            #pragma unroll
            for (int r = 0; r < 8; ++r) c[jt][r] *= SCALE;
        #pragma unroll
        for (int r = 0; r < 8; ++r) {
            float v = fmaxf(fmaxf(c[0][r], c[1][r]), fmaxf(c[2][r], c[3][r]));
            v = fmaxf(v, __shfl_xor(v, 1, 32));
            v = fmaxf(v, __shfl_xor(v, 2, 32));
            v = fmaxf(v, __shfl_xor(v, 4, 32));
            v = fmaxf(v, __shfl_xor(v, 8, 32));
            mnew[r]   = fmaxf(m_run[r], v);
            fscale[r] = __expf(m_run[r] - mnew[r]);
            m_run[r]  = mnew[r];
        }
        #pragma unroll
        for (int r = 0; r < 8; ++r) {
            float ps = 0.0f;
            #pragma unroll
            for (int jt = 0; jt < 4; ++jt) {
                float p = __expf(c[jt][r] - mnew[r]);
                ps += p;
                Pw[(h * 8 + r) * PP + jt * 16 + nl] = (_Float16)p;
            }
            ps += __shfl_xor(ps, 1, 32);
            ps += __shfl_xor(ps, 2, 32);
            ps += __shfl_xor(ps, 4, 32);
            ps += __shfl_xor(ps, 8, 32);
            l_run[r] = l_run[r] * fscale[r] + ps;
        }
        #pragma unroll
        for (int nt = 0; nt < 16; ++nt)
            #pragma unroll
            for (int r = 0; r < 8; ++r) o[nt][r] *= fscale[r];

        const _Float16* pr = Pw + nl * PP;
        v16h pa0 = ldfrag(pr + 8 * h,      pr + 16 + 8 * h);
        v16h pa1 = ldfrag(pr + 32 + 8 * h, pr + 48 + 8 * h);
        #pragma unroll
        for (int g = 0; g < 4; ++g) {
            v16h bv[8];
            #pragma unroll
            for (int i = 0; i < 4; ++i) {
                const _Float16* vb = &Vlds[((g * 4 + i) * 16 + nl) * VP];
                bv[2 * i]     = ldfrag(vb + 8 * h,      vb + 16 + 8 * h);
                bv[2 * i + 1] = ldfrag(vb + 32 + 8 * h, vb + 48 + 8 * h);
            }
            #pragma unroll
            for (int i = 0; i < 4; ++i) {
                const int nt = g * 4 + i;
                o[nt] = __builtin_amdgcn_wmma_f32_16x16x32_f16(
                            false, pa0, false, bv[2 * i], (short)0, o[nt], false, false);
                o[nt] = __builtin_amdgcn_wmma_f32_16x16x32_f16(
                            false, pa1, false, bv[2 * i + 1], (short)0, o[nt], false, false);
            }
        }
    }

    #pragma unroll
    for (int r = 0; r < 8; ++r) l_run[r] = 1.0f / l_run[r];
    #pragma unroll
    for (int nt = 0; nt < 16; ++nt) {
        #pragma unroll
        for (int r = 0; r < 8; ++r) {
            const int row = rb * TM + wv * 16 + h * 8 + r;
            Og[((size_t)(bh * TT + row)) * NN + nt * 16 + nl] = o[nt][r] * l_run[r];
        }
    }
}

// =====================================================================
extern "C" void kernel_launch(void* const* d_in, const int* in_sizes, int n_in,
                              void* d_out, int out_size, void* d_ws, size_t ws_size,
                              hipStream_t stream) {
    const float* Qg = (const float*)d_in[0];
    const float* Vg = (const float*)d_in[1];
    const float* Fg = (const float*)d_in[2];
    float* Og = (float*)d_out;
    (void)in_sizes; (void)n_in; (void)out_size;

    const size_t elems = (size_t)BH * TT * NN;          // 25,165,824
    const size_t need  = elems * 2 * 2;                 // QR + VT, f16
    if (ws_size >= need) {
        _Float16* QR = (_Float16*)d_ws;
        _Float16* VT = QR + elems;
        rope_prep_kernel  <<<dim3(BH * (TT / TS)), dim3(256), 0, stream>>>(Qg, Fg, QR);
        vtrans_prep_kernel<<<dim3(BH * (TT / TS)), dim3(256), 0, stream>>>(Vg, VT);
        fa2_kernel        <<<dim3(BH * (TT / TM)), dim3(256), 0, stream>>>(QR, VT, Og);
    } else {
        fa_fused_kernel   <<<dim3(BH * (TT / TM)), dim3(256), 0, stream>>>(Qg, Vg, Fg, Og);
    }
}
